// PointerNetworks_50440095924738
// MI455X (gfx1250) — matrix operand order repaired and checked by reading, not compiled, not run on hardware
//
#include <hip/hip_runtime.h>

#define HDIM 768
#define LDIM 512
#define BDIM 8
#define TDIM 32

typedef __attribute__((ext_vector_type(16))) __bf16 v16bf;
typedef __attribute__((ext_vector_type(8)))  float  v8f;

__device__ __forceinline__ float selu_f(float x) {
  const float kScale = 1.0507009873554805f;   // selu lambda
  const float kAS    = 1.7580993408473766f;   // lambda * alpha
  return x > 0.0f ? kScale * x : kAS * (__expf(x) - 1.0f);
}

// D[M x 768] = A[M x 768] * W[768 x 768]^T
// One 16x16 output tile per 32-thread (1 wave) block.
// Wave32 WMMA layouts (CDNA5 ISA):
//   A 16x32 bf16 : lane half h, row r=lane&15 -> elems 0..7 = K[k0+8h .. +7],
//                  elems 8..15 = K[k0+16+8h .. +7]
//   B 32x16 bf16 : lane half h, col r        -> elems 0..7 = K[k0+8h .. +7],
//                  elems 8..15 = K[k0+16+8h .. +7]  (B[k][n] = W[n][k])
//   D 16x16 f32  : vgpr i, lane -> row = 16*tileM + i + 8*half, col = 16*tileN + r
__global__ void __launch_bounds__(32)
gemm_abt_wmma(const float* __restrict__ A, const float* __restrict__ W,
              float* __restrict__ D) {
  const int tileM = blockIdx.x;
  const int tileN = blockIdx.y;
  const int lane  = threadIdx.x;
  const int half  = lane >> 4;
  const int r     = lane & 15;

  const float* arow = A + (size_t)(tileM * 16 + r) * HDIM;
  const float* wrow = W + (size_t)(tileN * 16 + r) * HDIM;

  v8f c = {0.f, 0.f, 0.f, 0.f, 0.f, 0.f, 0.f, 0.f};

  for (int k0 = 0; k0 < HDIM; k0 += 32) {
    float ab[16], bb[16];
    const float* pa0 = arow + k0 + half * 8;
    const float* pa1 = arow + k0 + 16 + half * 8;
    *(float4*)(ab + 0)  = *(const float4*)(pa0 + 0);
    *(float4*)(ab + 4)  = *(const float4*)(pa0 + 4);
    *(float4*)(ab + 8)  = *(const float4*)(pa1 + 0);
    *(float4*)(ab + 12) = *(const float4*)(pa1 + 4);
    const float* pb0 = wrow + k0 + half * 8;
    const float* pb1 = wrow + k0 + 16 + half * 8;
    *(float4*)(bb + 0)  = *(const float4*)(pb0 + 0);
    *(float4*)(bb + 4)  = *(const float4*)(pb0 + 4);
    *(float4*)(bb + 8)  = *(const float4*)(pb1 + 0);
    *(float4*)(bb + 12) = *(const float4*)(pb1 + 4);

    v16bf a, b;
#pragma unroll
    for (int i = 0; i < 16; ++i) {
      a[i] = (__bf16)ab[i];
      b[i] = (__bf16)bb[i];
    }
    // 8 args: (neg_a, A, neg_b, B, c_mod, C, reuse_a, reuse_b)
    c = __builtin_amdgcn_wmma_f32_16x16x32_bf16(false, a, false, b,
                                                (short)0, c, false, false);
  }

  float* drow = D + (size_t)(tileM * 16 + half * 8) * HDIM + tileN * 16 + r;
#pragma unroll
  for (int i = 0; i < 8; ++i) drow[(size_t)i * HDIM] = c[i];
}

// One block per (b, t). 512 threads = 16 waves.
// Fused: scores[l] = selu( sum_h v[h] * selu(WE[b,l,h] + WD[b,t,h]) )
// then masked log-softmax over l, out = mask ? logp : 0.
__global__ void __launch_bounds__(512)
pointer_softmax(const float* __restrict__ WE, const float* __restrict__ WD,
                const float* __restrict__ v, const int* __restrict__ starts,
                const int* __restrict__ lens, float* __restrict__ out) {
  const int bt  = blockIdx.x;    // b*T + t
  const int b   = bt >> 5;       // T = 32
  const int tid = threadIdx.x;

  __shared__ float wd[HDIM];
  __shared__ float vsh[HDIM];
  __shared__ float sc[LDIM];
  __shared__ float red[LDIM];

  for (int i = tid; i < HDIM; i += 512) {
    wd[i]  = WD[(size_t)bt * HDIM + i];
    vsh[i] = v[i];
  }
  __syncthreads();

  const int wave = tid >> 5;
  const int lane = tid & 31;
  for (int l = wave; l < LDIM; l += 16) {
    const float* we = WE + (size_t)(b * LDIM + l) * HDIM;
    float acc = 0.0f;
    for (int h = lane; h < HDIM; h += 32) {
      acc = fmaf(vsh[h], selu_f(we[h] + wd[h]), acc);
    }
#pragma unroll
    for (int off = 16; off > 0; off >>= 1) acc += __shfl_xor(acc, off, 32);
    if (lane == 0) sc[l] = selu_f(acc);
  }
  __syncthreads();

  const int start   = starts[bt];
  const int len     = lens[b];
  const bool inmask = (tid >= start) && (tid < len);
  const float s     = inmask ? sc[tid] : -1e30f;

  // block max
  red[tid] = s;
  __syncthreads();
#pragma unroll
  for (int st = 256; st > 0; st >>= 1) {
    if (tid < st) red[tid] = fmaxf(red[tid], red[tid + st]);
    __syncthreads();
  }
  const float mx = red[0];
  __syncthreads();

  // block sum of exp(s - mx)
  red[tid] = __expf(s - mx);
  __syncthreads();
#pragma unroll
  for (int st = 256; st > 0; st >>= 1) {
    if (tid < st) red[tid] += red[tid + st];
    __syncthreads();
  }
  const float lz = mx + __logf(red[0]);

  out[(size_t)bt * LDIM + tid] = inmask ? (sc[tid] - lz) : 0.0f;
}

extern "C" void kernel_launch(void* const* d_in, const int* in_sizes, int n_in,
                              void* d_out, int out_size, void* d_ws, size_t ws_size,
                              hipStream_t stream) {
  (void)in_sizes; (void)n_in; (void)out_size; (void)ws_size;

  const float* enc    = (const float*)d_in[0];  // [B, L, H]
  const float* dec    = (const float*)d_in[1];  // [B, T, H]
  const float* W1     = (const float*)d_in[2];  // [H, H]
  const float* W2     = (const float*)d_in[3];  // [H, H]
  const float* v      = (const float*)d_in[4];  // [H]
  const int*   starts = (const int*)d_in[5];    // [B, T]
  const int*   lens   = (const int*)d_in[6];    // [B]
  float*       out    = (float*)d_out;          // [B, T, L]

  float* WE = (float*)d_ws;                                   // 4096*768 f32
  float* WD = WE + (size_t)BDIM * LDIM * HDIM;                // 256*768 f32

  dim3 g1(BDIM * LDIM / 16, HDIM / 16);   // 256 x 48 tiles
  gemm_abt_wmma<<<g1, 32, 0, stream>>>(enc, W1, WE);

  dim3 g2(BDIM * TDIM / 16, HDIM / 16);   // 16 x 48 tiles
  gemm_abt_wmma<<<g2, 32, 0, stream>>>(dec, W2, WD);

  pointer_softmax<<<BDIM * TDIM, 512, 0, stream>>>(WE, WD, v, starts, lens, out);
}
